// MoELayer_62654982914897
// MI455X (gfx1250) — matrix operand order repaired and checked by reading, not compiled, not run on hardware
//
#include <hip/hip_runtime.h>
#include <hip/hip_bf16.h>
#include <hip/hip_fp16.h>

// ---------------- problem constants ----------------
#define NTOK 4096      // B*S = 2*2048 tokens
#define DMODEL 768
#define FFN 2048
#define NEXP 8
#define CAP 4096       // per-expert token-list capacity

// ---------------- tiling ----------------
#define BM 128         // token rows per block tile
#define BF 128         // F-chunk per block (gemm1 N, gemm2 K)
#define BN 128         // out-N chunk in gemm2
#define KSTEP 32
#define NKS1 (DMODEL / KSTEP)            // 24 K-steps in GEMM1
#define NSTEP2 ((DMODEL / BN) * (BF / KSTEP))  // 6 n-chunks * 4 k-steps = 24
#define NBLK_ROUTER 512

// ---------------- workspace layout (bytes) ----------------
#define WS_CNT   0                          // int cnt[8]
#define WS_LOAD  32                         // int load_cnt[8]
#define WS_IMP   64                         // float imp_part[512*8]
#define WS_Z     (64 + 512*8*4)             // float z_part[512]
#define WS_TOK   (WS_Z + 512*4)             // int tok_list[8*CAP]
#define WS_GATE  (WS_TOK + NEXP*CAP*4)      // float gate_list[8*CAP]
#define WS_END   (WS_GATE + NEXP*CAP*4)

typedef __attribute__((ext_vector_type(16))) __bf16 v16bf;
typedef __attribute__((ext_vector_type(8)))  float  v8f;

union FragU { v16bf v; uint4 q[2]; };

// A-fragment (16x32 bf16, MxK) from LDS: lanes 0-15 hold M=lane, K={0..7,16..23};
// lanes 16-31 hold M=lane-16, K={8..15,24..31}. Two contiguous 16B runs.
__device__ __forceinline__ v16bf ld_frag_a(const __bf16* base, int stride,
                                           int m0, int k0, int lane) {
  int r = m0 + (lane & 15);
  int k = k0 + ((lane & 16) ? 8 : 0);
  const __bf16* p = base + r * stride + k;
  FragU u;
  u.q[0] = *(const uint4*)(p);
  u.q[1] = *(const uint4*)(p + 16);
  return u.v;
}

// B-fragment (32x16 bf16, KxN) from LDS stored column-major [n][k]:
// lane holds column N=lane&15; lanes 0-15 K={0..7,16..23}, lanes 16-31 K={8..15,24..31}.
__device__ __forceinline__ v16bf ld_frag_b(const __bf16* base, int stride,
                                           int n0, int k0, int lane) {
  int n = n0 + (lane & 15);
  int k = k0 + ((lane & 16) ? 8 : 0);
  const __bf16* p = base + n * stride + k;
  FragU u;
  u.q[0] = *(const uint4*)(p);
  u.q[1] = *(const uint4*)(p + 16);
  return u.v;
}

// Build A fragment from 4 float4 (16 f32 in fragment element order).
__device__ __forceinline__ v16bf cvt_frag(float4 a0, float4 a1, float4 a2, float4 a3) {
  v16bf v;
  v[0]=(__bf16)a0.x; v[1]=(__bf16)a0.y; v[2]=(__bf16)a0.z; v[3]=(__bf16)a0.w;
  v[4]=(__bf16)a1.x; v[5]=(__bf16)a1.y; v[6]=(__bf16)a1.z; v[7]=(__bf16)a1.w;
  v[8]=(__bf16)a2.x; v[9]=(__bf16)a2.y; v[10]=(__bf16)a2.z; v[11]=(__bf16)a2.w;
  v[12]=(__bf16)a3.x; v[13]=(__bf16)a3.y; v[14]=(__bf16)a3.z; v[15]=(__bf16)a3.w;
  return v;
}

#define WMMA_BF16(A, B, C) __builtin_amdgcn_wmma_f32_16x16x32_bf16( \
    false, (A), false, (B), (short)0, (C), false, false)

// ================= router: logits, softmax, top-2, lists, loss partials ======
__global__ void __launch_bounds__(256)
moe_router(const float* __restrict__ x, const float* __restrict__ rw,
           int* __restrict__ cnt, int* __restrict__ loadc,
           float* __restrict__ imp_part, float* __restrict__ z_part,
           int* __restrict__ tok_list, float* __restrict__ gate_list) {
  __shared__ float rw_lds[DMODEL * NEXP];     // 24 KB
  __shared__ float imp_w[8][NEXP];
  __shared__ float z_w[8];
  const int tid = threadIdx.x, lane = tid & 31, wave = tid >> 5;

  for (int i = tid; i < DMODEL * NEXP; i += 256) rw_lds[i] = rw[i];
  __syncthreads();

  const int t = blockIdx.x * 8 + wave;        // one wave per token
  float lg[NEXP];
#pragma unroll
  for (int e = 0; e < NEXP; ++e) lg[e] = 0.f;
  const float* xp = x + (size_t)t * DMODEL;
  for (int i = 0; i < DMODEL / 32; ++i) {     // coalesced x reads
    int d = lane + 32 * i;
    float xv = xp[d];
#pragma unroll
    for (int e = 0; e < NEXP; ++e) lg[e] += xv * rw_lds[d * NEXP + e];
  }
#pragma unroll
  for (int e = 0; e < NEXP; ++e) {            // wave32 butterfly reduce
    float v = lg[e];
#pragma unroll
    for (int off = 16; off > 0; off >>= 1) v += __shfl_xor(v, off);
    lg[e] = v;
  }

  if (lane == 0) {
    float mx = lg[0];
#pragma unroll
    for (int e = 1; e < NEXP; ++e) mx = fmaxf(mx, lg[e]);
    float pr[NEXP], s = 0.f, zs = 0.f;
#pragma unroll
    for (int e = 0; e < NEXP; ++e) {
      zs += lg[e] * lg[e];
      pr[e] = __expf(lg[e] - mx);
      s += pr[e];
    }
    float inv = 1.f / s;
#pragma unroll
    for (int e = 0; e < NEXP; ++e) pr[e] *= inv;

    int i1 = 0; float p1 = pr[0];
#pragma unroll
    for (int e = 1; e < NEXP; ++e) if (pr[e] > p1) { p1 = pr[e]; i1 = e; }
    int i2 = (i1 == 0) ? 1 : 0; float p2 = pr[i2];
#pragma unroll
    for (int e = 0; e < NEXP; ++e)
      if (e != i1 && pr[e] > p2) { p2 = pr[e]; i2 = e; }
    float gsum = p1 + p2;
    float g1 = p1 / gsum, g2 = p2 / gsum;

    int s1 = atomicAdd(&cnt[i1], 1);
    tok_list[i1 * CAP + s1] = t;  gate_list[i1 * CAP + s1] = g1;
    int s2 = atomicAdd(&cnt[i2], 1);
    tok_list[i2 * CAP + s2] = t;  gate_list[i2 * CAP + s2] = g2;
    atomicAdd(&loadc[i1], 1);     // top-1 load count (integer -> deterministic)

#pragma unroll
    for (int e = 0; e < NEXP; ++e) imp_w[wave][e] = pr[e];
    z_w[wave] = zs;
  }
  __syncthreads();
  if (tid < NEXP) {               // deterministic per-block partials
    float s = 0.f;
    for (int w = 0; w < 8; ++w) s += imp_w[w][tid];
    imp_part[blockIdx.x * NEXP + tid] = s;
  }
  if (tid == NEXP) {
    float s = 0.f;
    for (int w = 0; w < 8; ++w) s += z_w[w];
    z_part[blockIdx.x] = s;
  }
}

// ================= finalize: aux & z losses (deterministic order) ============
__global__ void moe_finalize(const int* __restrict__ loadc,
                             const float* __restrict__ imp_part,
                             const float* __restrict__ z_part,
                             float* __restrict__ losses) {
  __shared__ float imp[NEXP];
  const int tid = threadIdx.x;
  if (tid < NEXP) {
    float s = 0.f;
    for (int b = 0; b < NBLK_ROUTER; ++b) s += imp_part[b * NEXP + tid];
    imp[tid] = s;
  }
  __syncthreads();
  if (tid == 0) {
    const float invn = 1.f / (float)NTOK;
    float aux = 0.f;
    for (int e = 0; e < NEXP; ++e)
      aux += (imp[e] * invn) * ((float)loadc[e] * invn);
    aux *= (float)NEXP * 0.01f;
    float zs = 0.f;
    for (int b = 0; b < NBLK_ROUTER; ++b) zs += z_part[b];
    losses[0] = aux;
    losses[1] = zs / (float)(NTOK * NEXP) * 0.001f;
  }
}

// ============ expert: fused silu(X@W1)*(X@W3) @ W2, gather + scatter-add =====
// Software-pipelined: global loads for slab k+1 issued while WMMAs consume
// slab k; A fragments built directly from global (per-wave private rows).
__global__ void __launch_bounds__(256)
moe_expert(const float* __restrict__ x, const float* __restrict__ w1,
           const float* __restrict__ w2, const float* __restrict__ w3,
           const int* __restrict__ cnt, const int* __restrict__ tok_list,
           const float* __restrict__ gate_list, float* __restrict__ out) {
  __shared__ __align__(16) __bf16 lds_b1[BF * KSTEP];  //  8 KB (W1 / W2 ping)
  __shared__ __align__(16) __bf16 lds_b3[BF * KSTEP];  //  8 KB (W3 / W2 pong)
  __shared__ __align__(16) __bf16 lds_h[BM * BF];      // 32 KB H tile (bf16)
  __shared__ int   tok_lds[BM];
  __shared__ float gate_lds[BM];

  const int e   = blockIdx.z;
  const int m0g = blockIdx.y * BM;      // token-tile origin in expert list
  const int f0  = blockIdx.x * BF;      // F-chunk origin
  if (m0g >= cnt[e]) return;            // tile fully past this expert's count

  const int tid = threadIdx.x, lane = tid & 31, wave = tid >> 5;
  const int strip = wave * 16;          // this wave's 16-token M strip

  for (int i = tid; i < BM; i += 256) {
    tok_lds[i]  = tok_list[e * CAP + m0g + i];   // padded slots are 0 (memset)
    gate_lds[i] = gate_list[e * CAP + m0g + i];  // padded gates are 0.0
  }
  __syncthreads();

  // Staging geometry (256 threads, 128-wide slabs, 16 regs/thread):
  //   column = tid & 127 (invariant), row(q) = 2q + (tid >> 7)
  const int scol = tid & 127;
  const int shi  = tid >> 7;
  const int lds_off = scol * KSTEP + shi;        // + 2q per reg

  // Per-lane A row (private to this wave's strip)
  const int atok = tok_lds[strip + (lane & 15)];
  const float* abase = x + (size_t)atok * DMODEL + ((lane & 16) ? 8 : 0);

  // ---- GEMM1: acc1 = X@W1[:,f0:BF], acc3 = X@W3[:,f0:BF] ----
  v8f acc1[8], acc3[8];
#pragma unroll
  for (int j = 0; j < 8; ++j) { acc1[j] = {}; acc3[j] = {}; }

  float w1reg[16], w3reg[16];
  float4 a0, a1, a2, a3;
  {  // prologue: fetch slab k=0 into registers
    const size_t wbase = ((size_t)e * DMODEL + shi) * FFN + f0 + scol;
#pragma unroll
    for (int q = 0; q < 16; ++q) {
      size_t gi = wbase + (size_t)(2 * q) * FFN;
      w1reg[q] = w1[gi];  w3reg[q] = w3[gi];
    }
    const float4* ap = (const float4*)abase;
    a0 = ap[0]; a1 = ap[1]; a2 = ap[4]; a3 = ap[5];
  }

  for (int step = 0; step < NKS1; ++step) {
    __syncthreads();                    // all waves done reading previous slab
#pragma unroll
    for (int q = 0; q < 16; ++q) {      // regs -> LDS (f32->bf16)
      lds_b1[lds_off + 2 * q] = (__bf16)w1reg[q];
      lds_b3[lds_off + 2 * q] = (__bf16)w3reg[q];
    }
    v16bf af = cvt_frag(a0, a1, a2, a3);
    __syncthreads();                    // slab visible to all waves
    if (step + 1 < NKS1) {              // prefetch slab k+1 (overlaps WMMAs)
      const int kn = (step + 1) * KSTEP;
      const size_t wbase = ((size_t)e * DMODEL + kn + shi) * FFN + f0 + scol;
#pragma unroll
      for (int q = 0; q < 16; ++q) {
        size_t gi = wbase + (size_t)(2 * q) * FFN;
        w1reg[q] = w1[gi];  w3reg[q] = w3[gi];
      }
      const float4* ap = (const float4*)(abase + kn);
      a0 = ap[0]; a1 = ap[1]; a2 = ap[4]; a3 = ap[5];
    }
#pragma unroll
    for (int j = 0; j < 8; ++j) {
      v16bf b1f = ld_frag_b(lds_b1, KSTEP, j * 16, 0, lane);
      v16bf b3f = ld_frag_b(lds_b3, KSTEP, j * 16, 0, lane);
      acc1[j] = WMMA_BF16(af, b1f, acc1[j]);
      acc3[j] = WMMA_BF16(af, b3f, acc3[j]);
    }
  }

  // ---- epilogue 1: H = silu(acc1) * acc3 -> LDS bf16 (C-layout mapping) ----
  __syncthreads();
#pragma unroll
  for (int j = 0; j < 8; ++j) {
#pragma unroll
    for (int r = 0; r < 8; ++r) {
      float a  = acc1[j][r];
      float hv = (a / (1.f + __expf(-a))) * acc3[j][r];
      int row = strip + ((lane & 16) ? 8 + r : r);
      int col = j * 16 + (lane & 15);
      lds_h[row * BF + col] = (__bf16)hv;
    }
  }

  // ---- GEMM2: (H tile) @ W2[f0:f0+BF, :], double-buffered W2 slabs --------
  // ping = lds_b1 (even steps), pong = lds_b3 (odd steps); no LDS-pointer
  // arrays (lld cannot relocate addrspacecast initializers).
  float w2reg[16];
  {  // prologue: stage step 0 (nc=0, kk=0) into ping buffer
    const size_t wbase = ((size_t)e * FFN + f0 + shi) * DMODEL + scol;
#pragma unroll
    for (int q = 0; q < 16; ++q) w2reg[q] = w2[wbase + (size_t)(2 * q) * DMODEL];
#pragma unroll
    for (int q = 0; q < 16; ++q) lds_b1[lds_off + 2 * q] = (__bf16)w2reg[q];
  }
  __syncthreads();                      // H tile + first W2 slab visible

  v8f acc[8];
#pragma unroll
  for (int j = 0; j < 8; ++j) acc[j] = {};

  for (int s = 0; s < NSTEP2; ++s) {
    const int nc = s >> 2, kk = (s & 3) * KSTEP;
    if (s + 1 < NSTEP2) {               // prefetch next W2 slab into regs
      const int nc1 = (s + 1) >> 2, kk1 = ((s + 1) & 3) * KSTEP;
      const size_t wbase =
          ((size_t)e * FFN + f0 + kk1 + shi) * DMODEL + nc1 * BN + scol;
#pragma unroll
      for (int q = 0; q < 16; ++q) w2reg[q] = w2[wbase + (size_t)(2 * q) * DMODEL];
    }
    v16bf af = ld_frag_a(lds_h, BF, strip, kk, lane);
    const __bf16* bcur = (s & 1) ? lds_b3 : lds_b1;
#pragma unroll
    for (int j = 0; j < 8; ++j) {
      v16bf bf2 = ld_frag_b(bcur, KSTEP, j * 16, 0, lane);
      acc[j] = WMMA_BF16(af, bf2, acc[j]);
    }
    if (s + 1 < NSTEP2) {               // store prefetched slab to other buffer
      __bf16* bnext = (s & 1) ? lds_b1 : lds_b3;
#pragma unroll
      for (int q = 0; q < 16; ++q) bnext[lds_off + 2 * q] = (__bf16)w2reg[q];
    }
    if ((s & 3) == 3) {                 // n-chunk done: gate-scale + scatter-add
      const int n0 = nc * BN;
#pragma unroll
      for (int j = 0; j < 8; ++j) {
#pragma unroll
        for (int r = 0; r < 8; ++r) {
          int row = strip + ((lane & 16) ? 8 + r : r);
          int col = n0 + j * 16 + (lane & 15);
          float v = acc[j][r] * gate_lds[row];     // 0.0 for padded rows
          unsafeAtomicAdd(&out[(size_t)tok_lds[row] * DMODEL + col], v);
        }
        acc[j] = {};
      }
    }
    __syncthreads();                    // buffer swap fence (1 barrier / step)
  }
}

// ================================ launcher ===================================
extern "C" void kernel_launch(void* const* d_in, const int* in_sizes, int n_in,
                              void* d_out, int out_size, void* d_ws, size_t ws_size,
                              hipStream_t stream) {
  const float* x  = (const float*)d_in[0];   // (2,2048,768)
  const float* rw = (const float*)d_in[1];   // (768,8)
  const float* w1 = (const float*)d_in[2];   // (8,768,2048)
  const float* w2 = (const float*)d_in[3];   // (8,2048,768)
  const float* w3 = (const float*)d_in[4];   // (8,768,2048)
  float* out = (float*)d_out;                // 4096*768 out + [aux, z]

  char* ws = (char*)d_ws;
  int*   cnt       = (int*)(ws + WS_CNT);
  int*   loadc     = (int*)(ws + WS_LOAD);
  float* imp_part  = (float*)(ws + WS_IMP);
  float* z_part    = (float*)(ws + WS_Z);
  int*   tok_list  = (int*)(ws + WS_TOK);
  float* gate_list = (float*)(ws + WS_GATE);

  // zero counters, lists (padding => token 0 / gate 0), and the output
  (void)hipMemsetAsync(d_ws, 0, (size_t)WS_END, stream);
  (void)hipMemsetAsync(d_out, 0, (size_t)out_size * sizeof(float), stream);

  moe_router<<<NBLK_ROUTER, 256, 0, stream>>>(x, rw, cnt, loadc, imp_part,
                                              z_part, tok_list, gate_list);
  moe_finalize<<<1, 64, 0, stream>>>(loadc, imp_part, z_part,
                                     out + (size_t)NTOK * DMODEL);
  moe_expert<<<dim3(FFN / BF, NTOK / BM, NEXP), 256, 0, stream>>>(
      x, w1, w2, w3, cnt, tok_list, gate_list, out);
}
